// SemanticMapSocialAttentionModel_66151086293054
// MI455X (gfx1250) — matrix operand order repaired and checked by reading, not compiled, not run on hardware
//
#include <hip/hip_runtime.h>
#include <hip/hip_bf16.h>
#include <math.h>

#define N_TGT 64
#define K_NB  48
#define TOBS  30
#define PREDN 15
#define EMB   64
#define NODE  128
#define EDGE  256
#define ATTD  64
#define CNN   512
#define M_NB  (N_TGT * K_NB)   // 3072

typedef __bf16 bf16_t;
typedef bf16_t v16bf __attribute__((ext_vector_type(16)));
typedef float  v8f   __attribute__((ext_vector_type(8)));
typedef int    v4i   __attribute__((ext_vector_type(4)));

union BfTile { v16bf v; uint4 q[2]; };

#if __has_builtin(__builtin_amdgcn_global_load_async_to_lds_b128)
#define HAVE_ASYNC_LDS 1
#else
#define HAVE_ASYNC_LDS 0
#endif

typedef __attribute__((address_space(1))) v4i* gv4i_ptr;   // global int4*
typedef __attribute__((address_space(3))) v4i* lv4i_ptr;   // LDS int4*

__device__ __forceinline__ gv4i_ptr to_global(const void* p) {
  return (gv4i_ptr)(unsigned long long)p;          // flat == global address value
}
__device__ __forceinline__ lv4i_ptr to_lds(const void* p) {
  // ISA 10.2: for LDS-aperture flat addresses, the LDS address is the low 32 bits
  return (lv4i_ptr)(unsigned int)(unsigned long long)p;
}

__device__ __forceinline__ float sigmoidf_(float x) { return 1.0f / (1.0f + __expf(-x)); }

// ---------------------------------------------------------------------------
// WMMA tile loaders (A/B in this file's contiguous k order, C/D per CDNA5 ISA layout)
// A (16x32 bf16, row-major src): lane L: m=L&15, h=L>>4
//   halfs 0-15 = K[h*16 .. h*16+15]  (contiguous, matching B)
// B (32x16 bf16 == W row-major [Nc x K]): lane L: n=L&15, h=L>>4
//   halfs 0-15 = K[h*16 .. h*16+15]  (contiguous along W row n)
// C/D (16x16 f32): col = L&15, row = (L>>4)*8 + r
// ---------------------------------------------------------------------------
__device__ __forceinline__ v16bf load_a16(const bf16_t* base, int ld) {
  int lane = threadIdx.x & 31;
  int m = lane & 15, h = lane >> 4;
  const bf16_t* p = base + m * ld + h * 16;
  BfTile t;
  t.q[0] = *(const uint4*)(p);
  t.q[1] = *(const uint4*)(p + 8);
  return t.v;
}
__device__ __forceinline__ v16bf load_b16(const bf16_t* base, int ld) {
  int lane = threadIdx.x & 31;
  int n = lane & 15, h = lane >> 4;
  const bf16_t* p = base + n * ld + h * 16;
  BfTile t;
  t.q[0] = *(const uint4*)(p);
  t.q[1] = *(const uint4*)(p + 8);
  return t.v;
}
__device__ __forceinline__ void store_c16(float* base, int ld, v8f c) {
  int lane = threadIdx.x & 31;
  int n = lane & 15, h = lane >> 4;
#pragma unroll
  for (int r = 0; r < 8; ++r) base[(h * 8 + r) * ld + n] = c[r];
}
__device__ __forceinline__ v8f wmma_bf16(v16bf a, v16bf b, v8f c) {
  return __builtin_amdgcn_wmma_f32_16x16x32_bf16(false, a, false, b, (short)0, c, false, false);
}

// ---------------------------------------------------------------------------
// Dual-source LSTM gate GEMM: G[M x Nc] = X[M x K1] @ W1[Nc x K1]^T
//                                       + H[M x K2] @ W2[Nc x K2]^T
// One wave -> 32x32 output block. blockDim = 256 (8 waves).
// ---------------------------------------------------------------------------
__global__ __launch_bounds__(256) void gemm_dual(
    const bf16_t* __restrict__ X, const bf16_t* __restrict__ W1,
    const bf16_t* __restrict__ H, const bf16_t* __restrict__ W2,
    float* __restrict__ G, int M, int Nc, int K1, int K2) {
  int wave = (int)((blockIdx.x * blockDim.x + threadIdx.x) >> 5);
  int tilesN = Nc >> 5;
  int wm = wave / tilesN, wn = wave - wm * tilesN;
  int m0 = wm * 32, n0 = wn * 32;
  if (m0 >= M) return;

  v8f acc00 = {}, acc01 = {}, acc10 = {}, acc11 = {};

  for (int k = 0; k < K1; k += 32) {
    v16bf a0 = load_a16(X + (size_t)m0 * K1 + k, K1);
    v16bf a1 = load_a16(X + (size_t)(m0 + 16) * K1 + k, K1);
    v16bf b0 = load_b16(W1 + (size_t)n0 * K1 + k, K1);
    v16bf b1 = load_b16(W1 + (size_t)(n0 + 16) * K1 + k, K1);
    acc00 = wmma_bf16(a0, b0, acc00);
    acc01 = wmma_bf16(a0, b1, acc01);
    acc10 = wmma_bf16(a1, b0, acc10);
    acc11 = wmma_bf16(a1, b1, acc11);
  }
  for (int k = 0; k < K2; k += 32) {
    __builtin_prefetch(H + (size_t)m0 * K2 + k + 64, 0, 1);
    v16bf a0 = load_a16(H + (size_t)m0 * K2 + k, K2);
    v16bf a1 = load_a16(H + (size_t)(m0 + 16) * K2 + k, K2);
    v16bf b0 = load_b16(W2 + (size_t)n0 * K2 + k, K2);
    v16bf b1 = load_b16(W2 + (size_t)(n0 + 16) * K2 + k, K2);
    acc00 = wmma_bf16(a0, b0, acc00);
    acc01 = wmma_bf16(a0, b1, acc01);
    acc10 = wmma_bf16(a1, b0, acc10);
    acc11 = wmma_bf16(a1, b1, acc11);
  }
  store_c16(G + (size_t)m0 * Nc + n0, Nc, acc00);
  store_c16(G + (size_t)m0 * Nc + n0 + 16, Nc, acc01);
  store_c16(G + (size_t)(m0 + 16) * Nc + n0, Nc, acc10);
  store_c16(G + (size_t)(m0 + 16) * Nc + n0 + 16, Nc, acc11);
}

// ---------------------------------------------------------------------------
// pred_out = [tht | img | Ht] @ W_pred^T + b_pred   (64 x 2)
// ---------------------------------------------------------------------------
__global__ __launch_bounds__(128) void pred_kernel(
    const float* __restrict__ tht, const float* __restrict__ img,
    const float* __restrict__ Ht, const float* __restrict__ Wp,
    const float* __restrict__ bp, float* __restrict__ pred) {
  int tid = threadIdx.x;            // 128 = 64 rows x 2 cols
  int n = tid >> 1, c = tid & 1;
  const float* w = Wp + c * (NODE + CNN + EDGE);
  float s = bp[c];
  for (int j = 0; j < NODE; ++j) s += tht[n * NODE + j] * w[j];
  for (int j = 0; j < CNN; ++j)  s += img[n * CNN + j] * w[NODE + j];
  for (int j = 0; j < EDGE; ++j) s += Ht[n * EDGE + j] * w[NODE + CNN + j];
  pred[tid] = s;
}

// ---------------------------------------------------------------------------
// Per-step state update, masks, gates; writes crel to output for t >= TOBS.
// One workgroup of 1024 threads.
// ---------------------------------------------------------------------------
__global__ __launch_bounds__(1024) void state_update(
    int t, const float* __restrict__ pred,
    const float* __restrict__ tabs, const float* __restrict__ trel,
    const float* __restrict__ tstep, const float* __restrict__ nabs_in,
    const float* __restrict__ nstep_in,
    const int* __restrict__ thist, const int* __restrict__ nhist,
    float* __restrict__ cabs, float* __restrict__ crel, float* __restrict__ cstep,
    float* __restrict__ nabs, float* __restrict__ nstep,
    float* __restrict__ tmask, float* __restrict__ nmask,
    float* __restrict__ flags, float* __restrict__ out) {
  __shared__ float bufT[1024];
  __shared__ float bufN[1024];
  int tid = threadIdx.x;
  bool is_obs = (t < TOBS);
  int tc = is_obs ? t : (TOBS - 1);

  float localT = 0.f, localN = 0.f;
  if (tid < N_TGT) {
    float m = is_obs ? ((thist[tid] > TOBS - t) ? 1.f : 0.f) : 1.f;
    tmask[tid] = m;
    localT = m;
  }
  for (int i = tid; i < M_NB; i += 1024) {
    float m = is_obs ? ((nhist[i] > TOBS - t) ? 1.f : 0.f) : 1.f;
    nmask[i] = m;
    localN += m;
  }
  bufT[tid] = localT;
  bufN[tid] = localN;
  __syncthreads();
  for (int s = 512; s > 0; s >>= 1) {
    if (tid < s) { bufT[tid] += bufT[tid + s]; bufN[tid] += bufN[tid + s]; }
    __syncthreads();
  }
  if (tid == 0) {
    float sT = bufT[0], sN = bufN[0];
    flags[0] = (sT > 0.f) ? 1.f : 0.f;                       // gate_t
    flags[1] = ((sT > 0.f) && (sN > 0.f)) ? 1.f : 0.f;       // gate_n
    flags[2] = sN;                                           // currN
  }

  // target position state (64 x 2)
  if (tid < 2 * N_TGT) {
    int n = tid >> 1, d = tid & 1;
    float p = pred[tid];
    float na = is_obs ? tabs[(n * TOBS + tc) * 2 + d] : (cabs[tid] + p);
    float nr = is_obs ? trel[(n * TOBS + tc) * 2 + d] : (crel[tid] + p);
    float ns = is_obs ? tstep[(n * TOBS + tc) * 2 + d] : p;
    cabs[tid] = na;
    crel[tid] = nr;
    cstep[tid] = ns;
    if (t >= TOBS) out[n * (PREDN * 2) + (t - TOBS) * 2 + d] = nr;
  }
  // nearby position state (3072 x 2): only refreshed during observation
  if (is_obs) {
    for (int i = tid; i < 2 * M_NB; i += 1024) {
      int pr = i >> 1, d = i & 1;       // pr = n*K + k
      nabs[i]  = nabs_in[(pr * TOBS + tc) * 2 + d];
      nstep[i] = nstep_in[(pr * TOBS + tc) * 2 + d];
    }
  }
}

// ---------------------------------------------------------------------------
// x = step2 @ W_disp^T + b_disp, emitted directly in bf16 for the WMMA GEMM.
// ---------------------------------------------------------------------------
__global__ __launch_bounds__(256) void disp_kernel(
    const float* __restrict__ s2, const float* __restrict__ Wd,
    const float* __restrict__ bd, bf16_t* __restrict__ xb, int rows) {
  int idx = blockIdx.x * blockDim.x + threadIdx.x;
  if (idx >= rows * EMB) return;
  int r = idx >> 6, e = idx & 63;
  float v = s2[2 * r] * Wd[2 * e] + s2[2 * r + 1] * Wd[2 * e + 1] + bd[e];
  xb[idx] = (bf16_t)v;
}

// ---------------------------------------------------------------------------
// Elementwise LSTM cell update with per-row mask and scalar gate.
// g layout: [rows x 4*width] as (i | f | g | o). Writes f32 h/c + bf16 h.
// ---------------------------------------------------------------------------
__global__ __launch_bounds__(256) void lstm_ew(
    const float* __restrict__ g, const float* __restrict__ bih,
    const float* __restrict__ bhh, float* __restrict__ h, float* __restrict__ c,
    bf16_t* __restrict__ hb, const float* __restrict__ mask,
    const float* __restrict__ gate, int rows, int width) {
  int idx = blockIdx.x * blockDim.x + threadIdx.x;
  if (idx >= rows * width) return;
  int r = idx / width, col = idx - r * width;
  const float* gr = g + (size_t)r * 4 * width;
  float gi = gr[col]             + bih[col]             + bhh[col];
  float gf = gr[width + col]     + bih[width + col]     + bhh[width + col];
  float gg = gr[2 * width + col] + bih[2 * width + col] + bhh[2 * width + col];
  float go = gr[3 * width + col] + bih[3 * width + col] + bhh[3 * width + col];
  float c2 = sigmoidf_(gf) * c[idx] + sigmoidf_(gi) * tanhf(gg);
  float h2 = sigmoidf_(go) * tanhf(c2);
  bool u = (mask[r] != 0.f) && (gate[0] != 0.f);
  float hn = u ? h2 : h[idx];
  float cn = u ? c2 : c[idx];
  h[idx] = hn;
  c[idx] = cn;
  hb[idx] = (bf16_t)hn;
}

// ---------------------------------------------------------------------------
// Masked social attention: one workgroup (256 threads) per target n.
// The 48x256 f32 slab of neighbor hidden states is staged into LDS with
// GLOBAL_LOAD_ASYNC_TO_LDS_B128 (ASYNCcnt path) when available.
// ---------------------------------------------------------------------------
__global__ __launch_bounds__(256) void attn_kernel(
    const float* __restrict__ crel, const float* __restrict__ cabs,
    const float* __restrict__ nabs, const float* __restrict__ nmask,
    const float* __restrict__ flags, const float* __restrict__ nht,
    const float* __restrict__ Wat, const float* __restrict__ bat,
    const float* __restrict__ Wan, const float* __restrict__ ban,
    float* __restrict__ Ht) {
  __shared__ __align__(16) float nh_s[K_NB * EDGE];   // 48 KB
  __shared__ float at[ATTD];
  __shared__ float sc[K_NB];
  __shared__ float w[K_NB];
  int n = blockIdx.x;
  int tid = threadIdx.x;

  const float* src = nht + (size_t)n * K_NB * EDGE;
#if HAVE_ASYNC_LDS
  // 3072 x 16B chunks; each of the 8 waves DMAs a disjoint slice.
  for (int c4 = tid; c4 < (K_NB * EDGE) / 4; c4 += 256) {
    __builtin_amdgcn_global_load_async_to_lds_b128(
        to_global(src + c4 * 4), to_lds(nh_s + c4 * 4), 0, 0);
  }
#else
  for (int i = tid; i < K_NB * EDGE; i += 256) nh_s[i] = src[i];
#endif

  if (tid < ATTD) {
    at[tid] = crel[n * 2] * Wat[tid * 2] + crel[n * 2 + 1] * Wat[tid * 2 + 1] + bat[tid];
  }

#if HAVE_ASYNC_LDS
#if __has_builtin(__builtin_amdgcn_s_wait_asynccnt)
  __builtin_amdgcn_s_wait_asynccnt(0);
#else
  asm volatile("s_wait_asynccnt 0" ::: "memory");
#endif
#endif
  __syncthreads();

  if (tid < K_NB) {
    float dx = nabs[(n * K_NB + tid) * 2]     - cabs[n * 2];
    float dy = nabs[(n * K_NB + tid) * 2 + 1] - cabs[n * 2 + 1];
    float s = 0.f;
    for (int a = 0; a < ATTD; ++a)
      s += (dx * Wan[a * 2] + dy * Wan[a * 2 + 1] + ban[a]) * at[a];
    s = s * flags[2] / 8.0f;              // * currN / sqrt(ATT)
    s *= nmask[n * K_NB + tid];
    sc[tid] = s;
  }
  __syncthreads();

  if (tid == 0) {
    float mx = sc[0];
    for (int k = 1; k < K_NB; ++k) mx = fmaxf(mx, sc[k]);
    float sum = 0.f;
    for (int k = 0; k < K_NB; ++k) {
      float e = __expf(sc[k] - mx) * nmask[n * K_NB + k];
      w[k] = e;
      sum += e;
    }
    float inv = 1.0f / (sum + 1e-6f);
    for (int k = 0; k < K_NB; ++k) w[k] *= inv;
  }
  __syncthreads();

  bool gate_n = (flags[1] != 0.f);
  for (int e = tid; e < EDGE; e += blockDim.x) {
    float acc = 0.f;
    for (int k = 0; k < K_NB; ++k)
      acc += w[k] * nh_s[k * EDGE + e];
    if (gate_n) Ht[n * EDGE + e] = acc;
  }
}

// ---------------------------------------------------------------------------
__global__ __launch_bounds__(256) void zero_f32(float* __restrict__ p, int n) {
  int i = blockIdx.x * blockDim.x + threadIdx.x;
  if (i < n) p[i] = 0.f;
}
__global__ __launch_bounds__(256) void cvt_bf16(const float* __restrict__ s,
                                                bf16_t* __restrict__ d, int n) {
  int i = blockIdx.x * blockDim.x + threadIdx.x;
  if (i < n) d[i] = (bf16_t)s[i];
}

// ---------------------------------------------------------------------------
extern "C" void kernel_launch(void* const* d_in, const int* in_sizes, int n_in,
                              void* d_out, int out_size, void* d_ws, size_t ws_size,
                              hipStream_t stream) {
  const float* img    = (const float*)d_in[0];
  const float* tabs   = (const float*)d_in[1];
  const float* trel   = (const float*)d_in[2];
  const float* tstep  = (const float*)d_in[3];
  const float* nabs_i = (const float*)d_in[4];
  // d_in[5] nearby_obs_pos_rel is unused by the forward pass
  const float* nstep_i = (const float*)d_in[6];
  const int*   thist  = (const int*)d_in[7];
  const int*   nhist  = (const int*)d_in[8];
  const float* Wd     = (const float*)d_in[9];
  const float* bd     = (const float*)d_in[10];
  const float* Wih_t  = (const float*)d_in[11];
  const float* Whh_t  = (const float*)d_in[12];
  const float* bih_t  = (const float*)d_in[13];
  const float* bhh_t  = (const float*)d_in[14];
  const float* Wih_n  = (const float*)d_in[15];
  const float* Whh_n  = (const float*)d_in[16];
  const float* bih_n  = (const float*)d_in[17];
  const float* bhh_n  = (const float*)d_in[18];
  const float* Wat    = (const float*)d_in[19];
  const float* bat    = (const float*)d_in[20];
  const float* Wan    = (const float*)d_in[21];
  const float* ban    = (const float*)d_in[22];
  const float* Wp     = (const float*)d_in[23];
  const float* bp     = (const float*)d_in[24];
  float* out = (float*)d_out;

  // ---- workspace layout (all offsets multiples of 16 floats) ----
  float* f = (float*)d_ws;
  size_t o = 0;
  auto alloc = [&](size_t n) { float* p = f + o; o += n; return p; };
  float* tht   = alloc(N_TGT * NODE);
  float* tct   = alloc(N_TGT * NODE);
  float* nht   = alloc((size_t)M_NB * EDGE);
  float* nct   = alloc((size_t)M_NB * EDGE);
  float* Ht    = alloc(N_TGT * EDGE);
  float* cabs  = alloc(128);
  float* crel  = alloc(128);
  float* cstep = alloc(128);
  float* nabs  = alloc(M_NB * 2);
  float* nstep = alloc(M_NB * 2);
  size_t zeroF = o;                       // state region to zero-init
  float* pred  = alloc(128);
  float* g_t   = alloc((size_t)N_TGT * 4 * NODE);
  float* g_n   = alloc((size_t)M_NB * 4 * EDGE);
  float* tmask = alloc(64);
  float* nmask = alloc(M_NB);
  float* flags = alloc(16);

  bf16_t* bfb = (bf16_t*)(f + o);
  size_t ob = 0;
  auto balloc = [&](size_t n) { bf16_t* p = bfb + ob; ob += n; return p; };
  bf16_t* thtb = balloc(N_TGT * NODE);
  bf16_t* nhtb = balloc((size_t)M_NB * EDGE);
  int zeroB = (int)((N_TGT * NODE + (size_t)M_NB * EDGE) / 2); // as floats
  bf16_t* x_t = balloc(N_TGT * EMB);
  bf16_t* x_n = balloc((size_t)M_NB * EMB);
  bf16_t* W1t = balloc(4 * NODE * EMB);
  bf16_t* W2t = balloc(4 * NODE * NODE);
  bf16_t* W1n = balloc(4 * EDGE * EMB);
  bf16_t* W2n = balloc((size_t)4 * EDGE * EDGE);

  // ---- one-time (per launch) init: zero state, bf16-convert weights ----
  zero_f32<<<(int)((zeroF + 255) / 256), 256, 0, stream>>>(f, (int)zeroF);
  zero_f32<<<(zeroB + 255) / 256, 256, 0, stream>>>((float*)thtb, zeroB);
  cvt_bf16<<<(4 * NODE * EMB + 255) / 256, 256, 0, stream>>>(Wih_t, W1t, 4 * NODE * EMB);
  cvt_bf16<<<(4 * NODE * NODE + 255) / 256, 256, 0, stream>>>(Whh_t, W2t, 4 * NODE * NODE);
  cvt_bf16<<<(4 * EDGE * EMB + 255) / 256, 256, 0, stream>>>(Wih_n, W1n, 4 * EDGE * EMB);
  cvt_bf16<<<(4 * EDGE * EDGE + 255) / 256, 256, 0, stream>>>(Whh_n, W2n, 4 * EDGE * EDGE);

  // ---- sequential scan: t = 1 .. TOBS+PREDN-1 ----
  for (int t = 1; t < TOBS + PREDN; ++t) {
    pred_kernel<<<1, 128, 0, stream>>>(tht, img, Ht, Wp, bp, pred);
    state_update<<<1, 1024, 0, stream>>>(t, pred, tabs, trel, tstep, nabs_i,
                                         nstep_i, thist, nhist, cabs, crel,
                                         cstep, nabs, nstep, tmask, nmask,
                                         flags, out);
    disp_kernel<<<(N_TGT * EMB + 255) / 256, 256, 0, stream>>>(cstep, Wd, bd, x_t, N_TGT);
    disp_kernel<<<(M_NB * EMB + 255) / 256, 256, 0, stream>>>(nstep, Wd, bd, x_n, M_NB);
    {
      int waves = (N_TGT / 32) * (4 * NODE / 32);   // 32 waves
      gemm_dual<<<waves / 8, 256, 0, stream>>>(x_t, W1t, thtb, W2t, g_t,
                                               N_TGT, 4 * NODE, EMB, NODE);
    }
    {
      int waves = (M_NB / 32) * (4 * EDGE / 32);    // 3072 waves
      gemm_dual<<<waves / 8, 256, 0, stream>>>(x_n, W1n, nhtb, W2n, g_n,
                                               M_NB, 4 * EDGE, EMB, EDGE);
    }
    lstm_ew<<<(N_TGT * NODE + 255) / 256, 256, 0, stream>>>(
        g_t, bih_t, bhh_t, tht, tct, thtb, tmask, flags + 0, N_TGT, NODE);
    lstm_ew<<<(M_NB * EDGE + 255) / 256, 256, 0, stream>>>(
        g_n, bih_n, bhh_n, nht, nct, nhtb, nmask, flags + 1, M_NB, EDGE);
    attn_kernel<<<N_TGT, 256, 0, stream>>>(crel, cabs, nabs, nmask, flags, nht,
                                           Wat, bat, Wan, ban, Ht);
  }
}
